// HANEncoder_15290083574224
// MI455X (gfx1250) — matrix operand order repaired and checked by reading, not compiled, not run on hardware
//
#include <hip/hip_runtime.h>
#include <math.h>

#define HID   128
#define HEADS 4
#define CDIM  32

typedef _Float16 h16;
typedef __attribute__((ext_vector_type(16))) _Float16 v16h;
typedef __attribute__((ext_vector_type(8)))  float    v8f;
typedef __attribute__((ext_vector_type(4)))  float    f32x4;

// ---------------- elementwise helpers ----------------

__global__ void cast_f16_kernel(const float* __restrict__ in, h16* __restrict__ out, long n) {
  long i = (long)blockIdx.x * blockDim.x + threadIdx.x;
  if (i < n) out[i] = (h16)in[i];
}

// W [Kin x 128] f32  ->  Wt [128 x Kin] f16 (transposed)
__global__ void wcast_kernel(const float* __restrict__ W, h16* __restrict__ Wt, int Kin) {
  int i = blockIdx.x * blockDim.x + threadIdx.x;
  if (i < Kin * HID) {
    int k = i / HID, n = i % HID;
    Wt[(size_t)n * Kin + k] = (h16)W[i];
  }
}

__global__ void fill_kernel(float* __restrict__ p, float v, long n) {
  long i = (long)blockIdx.x * blockDim.x + threadIdx.x;
  if (i < n) p[i] = v;
}

// ---------------- WMMA GEMM: C[M x 128] = A[M x K] @ Bt[128 x K]^T (+bias) ----------------
// One wave per 16x16 output tile; 8 waves/block cover the 128 output columns.
// A fragment (16x32 f16): lane m=l&15 holds row m; half kh=l>>4 holds
//   elements 0..7 -> K = kb + kh*8 + e, elements 8..15 -> K = kb + 16 + kh*8 + (e-8).
// B fragment (32x16 f16): lane n=l&15 holds column n; same K mapping as A.
template<int K>
__global__ __launch_bounds__(256) void gemm_kernel(const h16* __restrict__ A,
                                                   const h16* __restrict__ Bt,
                                                   const float* __restrict__ bias,
                                                   float* __restrict__ C, int M) {
  const int lane  = threadIdx.x & 31;
  const int wave  = threadIdx.x >> 5;
  const int tileM = blockIdx.x * 16;
  const int tileN = wave * 16;
  const int mn    = lane & 15;
  const int grp   = lane >> 4;

  const h16* arow = A  + (size_t)(tileM + mn) * K;
  const h16* brow = Bt + (size_t)(tileN + mn) * K;

  v8f acc = {0.f, 0.f, 0.f, 0.f, 0.f, 0.f, 0.f, 0.f};
  union Frag { v16h v; f32x4 f[2]; };

#pragma unroll
  for (int kb = 0; kb < K; kb += 32) {
    Frag a, b;
    a.f[0] = *(const f32x4*)(arow + kb + grp * 8);
    a.f[1] = *(const f32x4*)(arow + kb + 16 + grp * 8);
    b.f[0] = *(const f32x4*)(brow + kb + grp * 8);
    b.f[1] = *(const f32x4*)(brow + kb + 16 + grp * 8);
    acc = __builtin_amdgcn_wmma_f32_16x16x32_f16(false, a.v, false, b.v,
                                                 (short)0, acc, false, false);
  }

  const int  n     = lane & 15;
  const int  mbase = tileM + grp * 8;
  const float bv   = bias ? bias[tileN + n] : 0.f;
#pragma unroll
  for (int r = 0; r < 8; ++r)
    C[(size_t)(mbase + r) * HID + tileN + n] = acc[r] + bv;
}

// ---------------- attention projections: a[n,h] = sum_c H[n,h*32+c] * att[h*32+c] ----------------
__global__ void att_kernel(const float* __restrict__ H, const float* __restrict__ att,
                           float* __restrict__ out, int N) {
  int node = blockIdx.x * 8 + (threadIdx.x >> 5);
  int lane = threadIdx.x & 31;
  if (node >= N) return;
  const float* row = H + (size_t)node * HID;
  float p[HEADS];
#pragma unroll
  for (int h = 0; h < HEADS; ++h) p[h] = row[h * CDIM + lane] * att[h * CDIM + lane];
#pragma unroll
  for (int h = 0; h < HEADS; ++h)
#pragma unroll
    for (int off = 16; off > 0; off >>= 1) p[h] += __shfl_xor(p[h], off, 32);
  if (lane == 0) {
#pragma unroll
    for (int h = 0; h < HEADS; ++h) out[(size_t)node * HEADS + h] = p[h];
  }
}

// ---------------- edge passes ----------------

__device__ __forceinline__ void atomicMaxF32(float* addr, float v) {
  if (v >= 0.f) atomicMax((int*)addr, __float_as_int(v));
  else          atomicMin((unsigned int*)addr, __float_as_uint(v));
}

// pass 1: leaky-relu score + segment max over dst
__global__ void score_kernel(const int* __restrict__ src, const int* __restrict__ dst,
                             const float* __restrict__ a_src, const float* __restrict__ a_dst,
                             float* __restrict__ sbuf, float* __restrict__ mmax, long E4) {
  long i = (long)blockIdx.x * blockDim.x + threadIdx.x;
  if (i >= E4) return;
  long e = i >> 2; int h = i & 3;
  int s = src[e], d = dst[e];
  float sc = a_src[(size_t)s * 4 + h] + a_dst[(size_t)d * 4 + h];
  sc = sc > 0.f ? sc : 0.2f * sc;
  sbuf[i] = sc;
  atomicMaxF32(&mmax[(size_t)d * 4 + h], sc);
}

// pass 2: e = exp(s - m[dst]); denom[dst] += e
__global__ void expdenom_kernel(const int* __restrict__ dst, float* __restrict__ sbuf,
                                const float* __restrict__ mmax, float* __restrict__ denom, long E4) {
  long i = (long)blockIdx.x * blockDim.x + threadIdx.x;
  if (i >= E4) return;
  long e = i >> 2; int h = i & 3;
  int d = dst[e];
  float ex = __expf(sbuf[i] - mmax[(size_t)d * 4 + h]);
  sbuf[i] = ex;
  atomicAdd(&denom[(size_t)d * 4 + h], ex);
}

// pass 3: out[dst] += (e/denom[dst]) * Hsrc[src]   (one wave per edge, lane -> 4 channels)
__global__ void message_kernel(const int* __restrict__ src, const int* __restrict__ dst,
                               const float* __restrict__ sbuf, const float* __restrict__ denom,
                               const float* __restrict__ Hsrc, float* __restrict__ out, long E) {
  long e = (long)blockIdx.x * 8 + (threadIdx.x >> 5);
  if (e >= E) return;
  int lane = threadIdx.x & 31;
  int s = src[e], d = dst[e];
  int h = lane >> 3;  // channels [4*lane, 4*lane+3] all belong to head lane/8
  float alpha = sbuf[e * 4 + h] / (denom[(size_t)d * 4 + h] + 1e-16f);
  f32x4 x = *(const f32x4*)(Hsrc + (size_t)s * HID + lane * 4);
  float* o = out + (size_t)d * HID + lane * 4;
  atomicAdd(o + 0, alpha * x.x);
  atomicAdd(o + 1, alpha * x.y);
  atomicAdd(o + 2, alpha * x.z);
  atomicAdd(o + 3, alpha * x.w);
}

// ---------------- BatchNorm(+ReLU), training-mode biased var ----------------

__global__ void bn_stats_kernel(const float* __restrict__ X, float* __restrict__ sum,
                                float* __restrict__ sumsq, int N) {
  int col = threadIdx.x & 127;
  int row = blockIdx.x * 2 + (threadIdx.x >> 7);
  int stride = gridDim.x * 2;
  float s = 0.f, s2 = 0.f;
  for (int r = row; r < N; r += stride) {
    float v = X[(size_t)r * HID + col];
    s += v; s2 += v * v;
  }
  atomicAdd(&sum[col], s);
  atomicAdd(&sumsq[col], s2);
}

__global__ void bn_apply_kernel(const float* __restrict__ X, const float* __restrict__ sum,
                                const float* __restrict__ sumsq, const float* __restrict__ g,
                                const float* __restrict__ b, float* __restrict__ Y, int N) {
  long i = (long)blockIdx.x * blockDim.x + threadIdx.x;
  if (i >= (long)N * HID) return;
  int col = i & 127;
  float inv = 1.f / (float)N;
  float mu  = sum[col] * inv;
  float var = sumsq[col] * inv - mu * mu;
  float y = (X[i] - mu) * rsqrtf(var + 1e-5f) * g[col] + b[col];
  Y[i] = y > 0.f ? y : 0.f;
}

// ---------------- host orchestration ----------------

static inline char* carve(char*& cur, size_t bytes) {
  char* p = cur;
  cur += (bytes + 255) & ~(size_t)255;
  return p;
}

extern "C" void kernel_launch(void* const* d_in, const int* in_sizes, int n_in,
                              void* d_out, int out_size, void* d_ws, size_t ws_size,
                              hipStream_t stream) {
  (void)n_in; (void)out_size; (void)ws_size;

  const float* x_user = (const float*)d_in[0];
  const float* x_item = (const float*)d_in[1];
  const int*   e_uv   = (const int*)d_in[2];
  const int*   e_vu   = (const int*)d_in[3];
  const long E  = in_sizes[2] / 2;
  const int  NU = in_sizes[0] / 256;
  const int  NI = in_sizes[1] / 128;
  const int  NMAX = NU > NI ? NU : NI;

  const float* WlinU = (const float*)d_in[4];
  const float* blinU = (const float*)d_in[5];
  const float* WlinI = (const float*)d_in[6];
  const float* blinI = (const float*)d_in[7];

  // workspace carving
  char* cur = (char*)d_ws;
  float* xu      = (float*)carve(cur, (size_t)NU * HID * 4);
  float* xi      = (float*)carve(cur, (size_t)NI * HID * 4);
  h16*   xu_h    = (h16*)  carve(cur, (size_t)NU * HID * 2);
  h16*   xi_h    = (h16*)  carve(cur, (size_t)NI * HID * 2);
  h16*   xuser_h = (h16*)  carve(cur, (size_t)NU * 256 * 2);
  float* h_uv    = (float*)carve(cur, (size_t)NU * HID * 4);  // h_src for user->item
  float* h_vu    = (float*)carve(cur, (size_t)NI * HID * 4);  // h_src for item->user
  float* h_tmp   = (float*)carve(cur, (size_t)NMAX * HID * 4);
  float* outacc  = (float*)carve(cur, (size_t)NMAX * HID * 4);
  float* sbuf    = (float*)carve(cur, (size_t)E * 4 * 4);
  float* as_uv   = (float*)carve(cur, (size_t)NU * 4 * 4);
  float* ad_uv   = (float*)carve(cur, (size_t)NI * 4 * 4);
  float* as_vu   = (float*)carve(cur, (size_t)NI * 4 * 4);
  float* ad_vu   = (float*)carve(cur, (size_t)NU * 4 * 4);
  float* mmax    = (float*)carve(cur, (size_t)NMAX * 4 * 4);
  float* denom   = (float*)carve(cur, (size_t)NMAX * 4 * 4);
  float* bnsum   = (float*)carve(cur, 128 * 4);
  float* bnsq    = (float*)carve(cur, 128 * 4);
  h16*   WtU     = (h16*)  carve(cur, (size_t)256 * HID * 2);
  h16*   WtI     = (h16*)  carve(cur, (size_t)128 * HID * 2);
  h16*   WtG[4];
  for (int i = 0; i < 4; ++i) WtG[i] = (h16*)carve(cur, (size_t)128 * HID * 2);

  auto cast = [&](const float* in, h16* out, long n) {
    cast_f16_kernel<<<(unsigned)((n + 255) / 256), 256, 0, stream>>>(in, out, n);
  };
  auto fill = [&](float* p, float v, long n) {
    fill_kernel<<<(unsigned)((n + 255) / 256), 256, 0, stream>>>(p, v, n);
  };
  auto att = [&](const float* H, const float* a, float* o, int N) {
    att_kernel<<<(N + 7) / 8, 256, 0, stream>>>(H, a, o, N);
  };
  auto run_dir = [&](const int* src, const int* dst, const float* a_src, const float* a_dst,
                     const float* Hsrc, int Ndst) {
    long E4 = E * 4;
    fill(mmax,  -INFINITY, (long)Ndst * 4);
    fill(denom, 0.f,       (long)Ndst * 4);
    fill(outacc, 0.f,      (long)Ndst * HID);
    score_kernel  <<<(unsigned)((E4 + 255) / 256), 256, 0, stream>>>(src, dst, a_src, a_dst, sbuf, mmax, E4);
    expdenom_kernel<<<(unsigned)((E4 + 255) / 256), 256, 0, stream>>>(dst, sbuf, mmax, denom, E4);
    message_kernel<<<(unsigned)((E + 7) / 8),      256, 0, stream>>>(src, dst, sbuf, denom, Hsrc, outacc, E);
  };
  auto bn = [&](const float* X, const float* g, const float* b, float* Y, int N) {
    fill(bnsum, 0.f, 128);
    fill(bnsq,  0.f, 128);
    bn_stats_kernel<<<512, 256, 0, stream>>>(X, bnsum, bnsq, N);
    bn_apply_kernel<<<(unsigned)(((long)N * HID + 255) / 256), 256, 0, stream>>>(X, bnsum, bnsq, g, b, Y, N);
  };

  // weight prep (transpose + f16 cast)
  wcast_kernel<<<(256 * HID + 255) / 256, 256, 0, stream>>>(WlinU, WtU, 256);
  wcast_kernel<<<(128 * HID + 255) / 256, 256, 0, stream>>>(WlinI, WtI, 128);
  for (int l = 0; l < 2; ++l) {
    int base = 8 + l * 12;
    wcast_kernel<<<(128 * HID + 255) / 256, 256, 0, stream>>>((const float*)d_in[base + 0], WtG[2 * l + 0], 128);
    wcast_kernel<<<(128 * HID + 255) / 256, 256, 0, stream>>>((const float*)d_in[base + 4], WtG[2 * l + 1], 128);
  }

  // initial linears (WMMA)
  cast(x_user, xuser_h, (long)NU * 256);
  cast(x_item, xi_h,    (long)NI * 128);
  gemm_kernel<256><<<NU / 16, 256, 0, stream>>>(xuser_h, WtU, blinU, xu, NU);
  gemm_kernel<128><<<NI / 16, 256, 0, stream>>>(xi_h,    WtI, blinI, xi, NI);

  const int* uv_src = e_uv;     const int* uv_dst = e_uv + E;
  const int* vu_src = e_vu;     const int* vu_dst = e_vu + E;

  for (int l = 0; l < 2; ++l) {
    int base = 8 + l * 12;
    const float* att_src_uv = (const float*)d_in[base + 1];
    const float* att_dst_uv = (const float*)d_in[base + 2];
    const float* att_src_vu = (const float*)d_in[base + 5];
    const float* att_dst_vu = (const float*)d_in[base + 6];
    const float* gU = (const float*)d_in[base + 8];
    const float* bU = (const float*)d_in[base + 9];
    const float* gI = (const float*)d_in[base + 10];
    const float* bI = (const float*)d_in[base + 11];

    cast(xu, xu_h, (long)NU * HID);
    cast(xi, xi_h, (long)NI * HID);

    // projections through shared GAT linear (WMMA); GAT bias cancels inside BN.
    gemm_kernel<128><<<NU / 16, 256, 0, stream>>>(xu_h, WtG[2 * l + 0], nullptr, h_uv, NU);
    att(h_uv, att_src_uv, as_uv, NU);
    gemm_kernel<128><<<NI / 16, 256, 0, stream>>>(xi_h, WtG[2 * l + 0], nullptr, h_tmp, NI);
    att(h_tmp, att_dst_uv, ad_uv, NI);
    gemm_kernel<128><<<NI / 16, 256, 0, stream>>>(xi_h, WtG[2 * l + 1], nullptr, h_vu, NI);
    att(h_vu, att_src_vu, as_vu, NI);
    gemm_kernel<128><<<NU / 16, 256, 0, stream>>>(xu_h, WtG[2 * l + 1], nullptr, h_tmp, NU);
    att(h_tmp, att_dst_vu, ad_vu, NU);

    // user -> item direction, then BN(item)
    run_dir(uv_src, uv_dst, as_uv, ad_uv, h_uv, NI);
    float* xi_dst = (l == 1) ? ((float*)d_out + (size_t)NU * HID) : xi;
    bn(outacc, gI, bI, xi_dst, NI);

    // item -> user direction, then BN(user)
    run_dir(vu_src, vu_dst, as_vu, ad_vu, h_vu, NU);
    float* xu_dst = (l == 1) ? (float*)d_out : xu;
    bn(outacc, gU, bU, xu_dst, NU);
  }
}
